// TrittentionCube_47158740910637
// MI455X (gfx1250) — matrix operand order repaired and checked by reading, not compiled, not run on hardware
//
#include <hip/hip_runtime.h>

// ---------------------------------------------------------------------------
// TrittentionCube on MI455X (gfx1250, wave32, WMMA f32_16x16x32_f16).
// Fragment-native swizzled operand layouts (32B/lane b128 loads), async
// global->LDS staging (ASYNCcnt), strength-reduced fragment addressing.
// ---------------------------------------------------------------------------

typedef __attribute__((ext_vector_type(16))) _Float16 v16h;
typedef __attribute__((ext_vector_type(8)))  float    v8f;

#define T_SEQ   256
#define DMODEL  512
#define NH      8
#define HD      64
#define PROJ_N  2560
#define PC      32
#define NPC     8
#define NPART   (NH * 4 * NPC)
#define SLABSZ  (T_SEQ * 64)
#define FRAG    512            // elements per 32-lane fragment block

__device__ __forceinline__ v8f wmma32(v16h a, v16h b, v8f c) {
  return __builtin_amdgcn_wmma_f32_16x16x32_f16(false, a, false, b, (short)0, c,
                                                false, false);
}

// ---- CDNA5 async global->LDS (ASYNCcnt) ------------------------------------
__device__ __forceinline__ void async_copy_b128(unsigned lds_byte,
                                                const void* gaddr) {
  asm volatile("global_load_async_to_lds_b128 %0, %1, off"
               :: "v"(lds_byte), "v"((unsigned long long)(uintptr_t)gaddr)
               : "memory");
}
__device__ __forceinline__ void wait_asynccnt0() {
  asm volatile("s_wait_asynccnt 0x0" ::: "memory");
}

// ---- fragment-native offsets (contiguous k order: lane half h, element s -> k = 16h + s) -------------
__device__ __forceinline__ int afrag_off(int m, int k, int kblocks) {
  const int kt = k >> 5, kk = k & 31;
  const int hl = kk >> 4, t = kk & 15;
  const int lane = (m & 15) + (hl << 4);
  return ((((m >> 4) * kblocks + kt) << 5) + lane) * 16 + t;
}
__device__ __forceinline__ int bfrag_off(int k, int n, int nblocks) {
  const int lane = (n & 15) + (((k & 31) >> 4) << 4);
  return ((((k >> 5) * nblocks + (n >> 4)) << 5) + lane) * 16 + (k & 15);
}
__device__ __forceinline__ v16h ldv(const _Float16* p) {
  return *(const v16h*)p;
}
__device__ __forceinline__ v16h ld_afrag(const _Float16* base, int m0, int k0,
                                         int kblocks, int lane) {
  return ldv(base +
      (size_t)((((m0 >> 4) * kblocks + (k0 >> 5)) << 5) + lane) * 16);
}
__device__ __forceinline__ v16h ld_bfrag(const _Float16* base, int k0, int n0,
                                         int nblocks, int lane) {
  return ldv(base +
      (size_t)((((k0 >> 5) * nblocks + (n0 >> 4)) << 5) + lane) * 16);
}

// ---- C-tile stores (16x16 f32: lane -> N=lane&15, M = r+8*(lane>>4)) -------
__device__ __forceinline__ void st_tile_f32ln(float* dst, int ld, int m0,
                                              int n0, int lane, v8f c) {
  const int N = n0 + (lane & 15);
  const int Mb = m0 + 8 * (lane >> 4);
#pragma unroll
  for (int r = 0; r < 8; ++r) dst[(size_t)(Mb + r) * ld + N] = c[r];
}
__device__ __forceinline__ void st_tile_bfrag(_Float16* dst, int m0, int n0,
                                              int nblocks, int lane, v8f c) {
  const int col = n0 + (lane & 15);
  const int mb = m0 + 8 * (lane >> 4);
#pragma unroll
  for (int r = 0; r < 8; ++r)
    dst[bfrag_off(mb + r, col, nblocks)] = (_Float16)c[r];
}
__device__ __forceinline__ void st_tile_bfragT(_Float16* dst, int m0, int n0,
                                               int nblocks, int lane, v8f c) {
  const int col = n0 + (lane & 15);
  const int mb = m0 + 8 * (lane >> 4);
#pragma unroll
  for (int r = 0; r < 8; ++r)
    dst[bfrag_off(col, mb + r, nblocks)] = (_Float16)c[r];
}

// ---------------------------------------------------------------------------
// K0: converts into fragment-native layouts
// ---------------------------------------------------------------------------
__global__ void cvt_afrag_kernel(const float* __restrict__ src,
                                 _Float16* __restrict__ dst,
                                 int R, int W, int SW, int n) {
  const int i = blockIdx.x * blockDim.x + threadIdx.x;
  if (i >= n) return;
  const int m = i / W, c = i % W;
  const int slab = c / SW, k = c % SW;
  dst[(size_t)slab * R * SW + afrag_off(m, k, SW >> 5)] = (_Float16)src[i];
}
__global__ void cvt_bfrag_kernel(const float* __restrict__ src,
                                 _Float16* __restrict__ dst,
                                 int K, int N, int n) {
  const int i = blockIdx.x * blockDim.x + threadIdx.x;
  if (i >= n) return;
  const int slabSz = K * N;
  const int slab = i / slabSz, r = i % slabSz;
  dst[(size_t)slab * slabSz + bfrag_off(r / N, r % N, N >> 4)] =
      (_Float16)src[i];
}

// ---------------------------------------------------------------------------
// K1: proj = x @ W_kkqvv + b -> afragged 64-col head slabs.
// Wave tile 32x64; strength-reduced fragment pointers.
// ---------------------------------------------------------------------------
__global__ __launch_bounds__(128) void gemm_proj_kernel(
    const _Float16* __restrict__ Aaf,   // kblocks = 16
    const _Float16* __restrict__ Bbf,   // nblocks = 160
    const float* __restrict__ bias,
    _Float16* __restrict__ Caf) {
  const int wave = threadIdx.x >> 5, lane = threadIdx.x & 31;
  const int m0 = blockIdx.y * 128 + wave * 32;
  const int n0 = blockIdx.x * 64;
  const _Float16* ap = Aaf + ((size_t)(m0 >> 4) * 16 * 32 + lane) * 16;
  const _Float16* bp = Bbf + ((size_t)(n0 >> 4) * 32 + lane) * 16;
  v8f acc[2][4] = {};
  for (int k0 = 0; k0 < DMODEL; k0 += 32) {
    v16h a0 = ldv(ap);
    v16h a1 = ldv(ap + 16 * FRAG);   // next m-tile (16 k-blocks away)
#pragma unroll
    for (int nn = 0; nn < 4; ++nn) {
      v16h bf = ldv(bp + nn * FRAG);
      acc[0][nn] = wmma32(a0, bf, acc[0][nn]);
      acc[1][nn] = wmma32(a1, bf, acc[1][nn]);
    }
    ap += FRAG;
    bp += 160 * FRAG;
  }
#pragma unroll
  for (int mm = 0; mm < 2; ++mm) {
    const int mb = m0 + mm * 16 + 8 * (lane >> 4);
#pragma unroll
    for (int nn = 0; nn < 4; ++nn) {
      const int col = n0 + nn * 16 + (lane & 15);
      const float bv = bias[col];
      _Float16* dst = Caf + (size_t)(col >> 6) * SLABSZ;
      const int kc = col & 63;
#pragma unroll
      for (int r = 0; r < 8; ++r)
        dst[afrag_off(mb + r, kc, 2)] = (_Float16)(acc[mm][nn][r] + bv);
    }
  }
}

// ---------------------------------------------------------------------------
// K2: per head O[p, j*64+i] = proj_slab @ W ; row p stored as bfrag 64x64.
// ---------------------------------------------------------------------------
__global__ __launch_bounds__(128) void gemm_kw_kernel(
    const _Float16* __restrict__ projAf, int slab0,
    const _Float16* __restrict__ Wbf,    // nblocks = 256
    _Float16* __restrict__ Obf) {
  const int h = blockIdx.z;
  const _Float16* Aaf = projAf + (size_t)(slab0 + h) * SLABSZ;
  const _Float16* B = Wbf + (size_t)h * 64 * 4096;
  _Float16* O = Obf + (size_t)h * T_SEQ * 4096;
  const int wave = threadIdx.x >> 5, lane = threadIdx.x & 31;
  const int m0 = blockIdx.y * 128 + wave * 32;
  const int n0 = blockIdx.x * 64;
  const _Float16* ap = Aaf + ((size_t)(m0 >> 4) * 2 * 32 + lane) * 16;
  const _Float16* bp = B + ((size_t)(n0 >> 4) * 32 + lane) * 16;
  v8f acc[2][4] = {};
#pragma unroll
  for (int k0 = 0; k0 < 64; k0 += 32) {
    v16h a0 = ldv(ap);
    v16h a1 = ldv(ap + 2 * FRAG);
#pragma unroll
    for (int nn = 0; nn < 4; ++nn) {
      v16h bf = ldv(bp + nn * FRAG);
      acc[0][nn] = wmma32(a0, bf, acc[0][nn]);
      acc[1][nn] = wmma32(a1, bf, acc[1][nn]);
    }
    ap += FRAG;
    bp += 256 * FRAG;
  }
#pragma unroll
  for (int mm = 0; mm < 2; ++mm) {
    const int mb = m0 + mm * 16 + 8 * (lane >> 4);
#pragma unroll
    for (int nn = 0; nn < 4; ++nn) {
      const int col = n0 + nn * 16 + (lane & 15);
      const int j = col >> 6, i = col & 63;
#pragma unroll
      for (int r = 0; r < 8; ++r)
        O[(size_t)(mb + r) * 4096 + bfrag_off(j, i, 4)] =
            (_Float16)acc[mm][nn][r];
    }
  }
}

// ---------------------------------------------------------------------------
// K3: fused flash attention over the 2D (p,t) key space.
// ---------------------------------------------------------------------------
__global__ __launch_bounds__(256) void trittention_flash_kernel(
    const _Float16* __restrict__ projh,
    const _Float16* __restrict__ Abuf,
    const _Float16* __restrict__ Vbuf,
    float* __restrict__ pm, float* __restrict__ pl,
    float* __restrict__ pacc) {
  const int h  = blockIdx.z;
  const int qt = blockIdx.y;
  const int pc = blockIdx.x;
  const int tid  = threadIdx.x;
  const int lane = tid & 31;
  const int wave = tid >> 5;
  const int mi  = wave >> 1;
  const int ni0 = (wave & 1) * 2;
  const int q0 = qt * 64;

  __shared__ __align__(32) _Float16 sAp[64 * 64];
  __shared__ __align__(32) _Float16 sVTp[64 * 64];
  __shared__ __align__(32) _Float16 sWork[64 * 64];
  __shared__ __align__(32) _Float16 sP[64 * 64];
  __shared__ float sS[64 * 64];
  __shared__ float sM[64], sL[64], sScale[64];

  if (tid < 64) { sM[tid] = -__builtin_inff(); sL[tid] = 0.0f; }
  v8f acc0 = {}, acc1 = {};

  const _Float16* k2p = projh + (size_t)(8 + h) * SLABSZ + (size_t)lane * 16;
  const _Float16* v2p = projh + (size_t)(32 + h) * SLABSZ + (size_t)lane * 16;
  const _Float16* Aph = Abuf + (size_t)h * T_SEQ * 4096;
  const _Float16* Vph = Vbuf + (size_t)h * T_SEQ * 4096;
  const unsigned ldsA = (unsigned)(uintptr_t)sAp;
  const unsigned ldsV = (unsigned)(uintptr_t)sVTp;

  // q-row A-fragments are invariant over the whole kernel: hoist to registers
  const _Float16* qp = projh + (size_t)(16 + h) * SLABSZ;
  const v16h qa0 = ld_afrag(qp, q0 + mi * 16, 0, 2, lane);
  const v16h qa1 = ld_afrag(qp, q0 + mi * 16, 32, 2, lane);
  __syncthreads();

  for (int pp = 0; pp < PC; ++pp) {
    const int p = pc * PC + pp;
    {   // async-stage the two 8KB fragment-ready rows (ASYNCcnt path)
      const char* gA = (const char*)(Aph + (size_t)p * 4096);
      const char* gV = (const char*)(Vph + (size_t)p * 4096);
      const unsigned o = tid * 16;
      async_copy_b128(ldsA + o,        gA + o);
      async_copy_b128(ldsA + o + 4096, gA + o + 4096);
      async_copy_b128(ldsV + o,        gV + o);
      async_copy_b128(ldsV + o + 4096, gV + o + 4096);
      if (pp + 1 < PC) {
        __builtin_prefetch(Aph + (size_t)(p + 1) * 4096, 0, 1);
        __builtin_prefetch(Vph + (size_t)(p + 1) * 4096, 0, 1);
      }
      wait_asynccnt0();
    }
    __syncthreads();

    for (int tc = 0; tc < 4; ++tc) {
      const int t0 = tc * 64;
      const size_t rowblk = (size_t)((t0 >> 4) + mi) * (2 * FRAG);
      // ---- s1[t,i] = k2[t0.., :] @ A_p ; stored TRANSPOSED (k=i, n=t) ------
      {
        v8f c0 = {}, c1 = {};
        v16h a0 = ldv(k2p + rowblk);
        v16h a1 = ldv(k2p + rowblk + FRAG);
        v16h b0 = ld_bfrag(sAp, 0, (ni0 + 0) * 16, 4, lane);
        v16h b1 = ld_bfrag(sAp, 0, (ni0 + 1) * 16, 4, lane);
        c0 = wmma32(a0, b0, c0);
        c1 = wmma32(a0, b1, c1);
        b0 = ld_bfrag(sAp, 32, (ni0 + 0) * 16, 4, lane);
        b1 = ld_bfrag(sAp, 32, (ni0 + 1) * 16, 4, lane);
        c0 = wmma32(a1, b0, c0);
        c1 = wmma32(a1, b1, c1);
        st_tile_bfragT(sWork, mi * 16, (ni0 + 0) * 16, 4, lane, c0);
        st_tile_bfragT(sWork, mi * 16, (ni0 + 1) * 16, 4, lane, c1);
      }
      __syncthreads();
      // ---- S[q,t] = qmat @ s1^T --------------------------------------------
      {
        v8f c0 = {}, c1 = {};
        v16h b0 = ld_bfrag(sWork, 0, (ni0 + 0) * 16, 4, lane);
        v16h b1 = ld_bfrag(sWork, 0, (ni0 + 1) * 16, 4, lane);
        c0 = wmma32(qa0, b0, c0);
        c1 = wmma32(qa0, b1, c1);
        b0 = ld_bfrag(sWork, 32, (ni0 + 0) * 16, 4, lane);
        b1 = ld_bfrag(sWork, 32, (ni0 + 1) * 16, 4, lane);
        c0 = wmma32(qa1, b0, c0);
        c1 = wmma32(qa1, b1, c1);
        st_tile_f32ln(sS, 64, mi * 16, (ni0 + 0) * 16, lane, c0);
        st_tile_f32ln(sS, 64, mi * 16, (ni0 + 1) * 16, lane, c1);
      }
      __syncthreads();
      // ---- masked online softmax; P written in afrag layout ----------------
      {
        const int row = tid >> 2, seg = tid & 3;
        const int qg = q0 + row;
        const float mold = sM[row];
        float vmax = -__builtin_inff();
        float vals[16];
#pragma unroll
        for (int c = 0; c < 16; ++c) {
          const int col = seg * 16 + c;
          const bool valid = (p <= qg) && (t0 + col <= qg);
          const float v = valid ? sS[row * 64 + col] * (1.0f / 64.0f)
                                : -15625.0f;
          vals[c] = v;
          vmax = fmaxf(vmax, v);
        }
        vmax = fmaxf(vmax, __shfl_xor(vmax, 1));
        vmax = fmaxf(vmax, __shfl_xor(vmax, 2));
        const float mnew = fmaxf(mold, vmax);
        float lsum = 0.0f;
#pragma unroll
        for (int c = 0; c < 16; ++c) {
          const float e = __expf(vals[c] - mnew);
          lsum += e;
          sP[afrag_off(row, seg * 16 + c, 2)] = (_Float16)e;
        }
        lsum += __shfl_xor(lsum, 1);
        lsum += __shfl_xor(lsum, 2);
        if (seg == 0) {
          const float f = __expf(mold - mnew);
          sScale[row] = f;
          sM[row] = mnew;
          sL[row] = sL[row] * f + lsum;
        }
      }
      // ---- V[s,e] = v2[t0.., :] @ VT_p ; stored bfrag (k=s, n=e) -----------
      {
        v8f c0 = {}, c1 = {};
        v16h a0 = ldv(v2p + rowblk);
        v16h a1 = ldv(v2p + rowblk + FRAG);
        v16h b0 = ld_bfrag(sVTp, 0, (ni0 + 0) * 16, 4, lane);
        v16h b1 = ld_bfrag(sVTp, 0, (ni0 + 1) * 16, 4, lane);
        c0 = wmma32(a0, b0, c0);
        c1 = wmma32(a0, b1, c1);
        b0 = ld_bfrag(sVTp, 32, (ni0 + 0) * 16, 4, lane);
        b1 = ld_bfrag(sVTp, 32, (ni0 + 1) * 16, 4, lane);
        c0 = wmma32(a1, b0, c0);
        c1 = wmma32(a1, b1, c1);
        st_tile_bfrag(sWork, mi * 16, (ni0 + 0) * 16, 4, lane, c0);
        st_tile_bfrag(sWork, mi * 16, (ni0 + 1) * 16, 4, lane, c1);
      }
      __syncthreads();
      // ---- acc = acc * scale + P @ V ---------------------------------------
      {
        const int Mb = mi * 16 + 8 * (lane >> 4);
#pragma unroll
        for (int r = 0; r < 8; ++r) {
          const float f = sScale[Mb + r];
          acc0[r] *= f;
          acc1[r] *= f;
        }
        v16h a0 = ld_afrag(sP, mi * 16, 0, 2, lane);
        v16h a1 = ld_afrag(sP, mi * 16, 32, 2, lane);
        v16h b0 = ld_bfrag(sWork, 0, (ni0 + 0) * 16, 4, lane);
        v16h b1 = ld_bfrag(sWork, 0, (ni0 + 1) * 16, 4, lane);
        acc0 = wmma32(a0, b0, acc0);
        acc1 = wmma32(a0, b1, acc1);
        b0 = ld_bfrag(sWork, 32, (ni0 + 0) * 16, 4, lane);
        b1 = ld_bfrag(sWork, 32, (ni0 + 1) * 16, 4, lane);
        acc0 = wmma32(a1, b0, acc0);
        acc1 = wmma32(a1, b1, acc1);
      }
      __syncthreads();
    }
  }

  const int part = (h * 4 + qt) * NPC + pc;
  if (tid < 64) {
    pm[part * 64 + tid] = sM[tid];
    pl[part * 64 + tid] = sL[tid];
  }
  float* accout = pacc + (size_t)part * 4096;
  st_tile_f32ln(accout, 64, mi * 16, (ni0 + 0) * 16, lane, acc0);
  st_tile_f32ln(accout, 64, mi * 16, (ni0 + 1) * 16, lane, acc1);
}

// ---------------------------------------------------------------------------
// K4: LSE merge of NPC partials -> z (afragged f16 [256 x 512])
// ---------------------------------------------------------------------------
__global__ void merge_kernel(const float* __restrict__ pm,
                             const float* __restrict__ pl,
                             const float* __restrict__ pacc,
                             _Float16* __restrict__ zaf) {
  const int g = blockIdx.x;
  const int h = g >> 2, qt = g & 3;
  const int row = threadIdx.x;
  float M = -__builtin_inff();
  for (int i = 0; i < NPC; ++i) M = fmaxf(M, pm[(g * NPC + i) * 64 + row]);
  float L = 0.0f;
  float w[NPC];
  for (int i = 0; i < NPC; ++i) {
    const float f = __expf(pm[(g * NPC + i) * 64 + row] - M);
    w[i] = f;
    L += pl[(g * NPC + i) * 64 + row] * f;
  }
  const float invL = 1.0f / L;
  const int m = qt * 64 + row;
  for (int e = 0; e < HD; ++e) {
    float s = 0.0f;
    for (int i = 0; i < NPC; ++i)
      s += pacc[(size_t)(g * NPC + i) * 4096 + row * 64 + e] * w[i];
    zaf[afrag_off(m, h * HD + e, 16)] = (_Float16)(s * invL);
  }
}

// ---------------------------------------------------------------------------
// K5: out = z @ W_out + b_out -> f32 row-major d_out
// ---------------------------------------------------------------------------
__global__ __launch_bounds__(128) void gemm_out_kernel(
    const _Float16* __restrict__ Zaf,   // kblocks = 16
    const _Float16* __restrict__ Wbf,   // nblocks = 32
    const float* __restrict__ bias,
    float* __restrict__ out) {
  const int wave = threadIdx.x >> 5, lane = threadIdx.x & 31;
  const int m0 = blockIdx.y * 128 + wave * 32;
  const int n0 = blockIdx.x * 64;
  const _Float16* ap = Zaf + ((size_t)(m0 >> 4) * 16 * 32 + lane) * 16;
  const _Float16* bp = Wbf + ((size_t)(n0 >> 4) * 32 + lane) * 16;
  v8f acc[2][4] = {};
  for (int k0 = 0; k0 < DMODEL; k0 += 32) {
    v16h a0 = ldv(ap);
    v16h a1 = ldv(ap + 16 * FRAG);
#pragma unroll
    for (int nn = 0; nn < 4; ++nn) {
      v16h bf = ldv(bp + nn * FRAG);
      acc[0][nn] = wmma32(a0, bf, acc[0][nn]);
      acc[1][nn] = wmma32(a1, bf, acc[1][nn]);
    }
    ap += FRAG;
    bp += 32 * FRAG;
  }
#pragma unroll
  for (int mm = 0; mm < 2; ++mm) {
    const int mb = m0 + mm * 16 + 8 * (lane >> 4);
#pragma unroll
    for (int nn = 0; nn < 4; ++nn) {
      const int col = n0 + nn * 16 + (lane & 15);
      const float bv = bias[col];
#pragma unroll
      for (int r = 0; r < 8; ++r)
        out[(size_t)(mb + r) * DMODEL + col] = acc[mm][nn][r] + bv;
    }
  }
}

// ---------------------------------------------------------------------------
extern "C" void kernel_launch(void* const* d_in, const int* in_sizes, int n_in,
                              void* d_out, int out_size, void* d_ws,
                              size_t ws_size, hipStream_t stream) {
  (void)in_sizes; (void)n_in; (void)out_size; (void)ws_size;
  const float* x    = (const float*)d_in[0];
  const float* Wk   = (const float*)d_in[1];
  const float* bk   = (const float*)d_in[2];
  const float* WKq  = (const float*)d_in[3];
  const float* WVq  = (const float*)d_in[4];
  const float* Wout = (const float*)d_in[5];
  const float* bout = (const float*)d_in[6];
  float* out = (float*)d_out;

  char* ws = (char*)d_ws;
  size_t off = 0;
  auto alloc = [&](size_t bytes) -> void* {
    void* p = ws + off;
    off = (off + bytes + 255) & ~(size_t)255;
    return p;
  };
  _Float16* xh    = (_Float16*)alloc((size_t)T_SEQ * DMODEL * 2);
  _Float16* Wkh   = (_Float16*)alloc((size_t)DMODEL * PROJ_N * 2);
  _Float16* WKqh  = (_Float16*)alloc((size_t)NH * 64 * 4096 * 2);
  _Float16* WVqh  = (_Float16*)alloc((size_t)NH * 64 * 4096 * 2);
  _Float16* Wouth = (_Float16*)alloc((size_t)DMODEL * DMODEL * 2);
  _Float16* projh = (_Float16*)alloc((size_t)T_SEQ * PROJ_N * 2);
  _Float16* Abuf  = (_Float16*)alloc((size_t)NH * T_SEQ * 4096 * 2);
  _Float16* Vbuf  = (_Float16*)alloc((size_t)NH * T_SEQ * 4096 * 2);
  _Float16* zh    = (_Float16*)alloc((size_t)T_SEQ * DMODEL * 2);
  float*    pm    = (float*)alloc((size_t)NPART * 64 * 4);
  float*    pl    = (float*)alloc((size_t)NPART * 64 * 4);
  float*    pacc  = (float*)alloc((size_t)NPART * 4096 * 4);

  cvt_afrag_kernel<<<131072 / 256, 256, 0, stream>>>(x, xh, 256, 512, 512,
                                                     131072);
  cvt_bfrag_kernel<<<1310720 / 256, 256, 0, stream>>>(Wk, Wkh, 512, 2560,
                                                      1310720);
  cvt_bfrag_kernel<<<2097152 / 256, 256, 0, stream>>>(WKq, WKqh, 64, 4096,
                                                      2097152);
  cvt_bfrag_kernel<<<2097152 / 256, 256, 0, stream>>>(WVq, WVqh, 64, 4096,
                                                      2097152);
  cvt_bfrag_kernel<<<262144 / 256, 256, 0, stream>>>(Wout, Wouth, 512, 512,
                                                     262144);

  gemm_proj_kernel<<<dim3(PROJ_N / 64, T_SEQ / 128, 1), 128, 0, stream>>>(
      xh, Wkh, bk, projh);

  gemm_kw_kernel<<<dim3(64, T_SEQ / 128, NH), 128, 0, stream>>>(projh, 0, WKqh,
                                                                Abuf);
  gemm_kw_kernel<<<dim3(64, T_SEQ / 128, NH), 128, 0, stream>>>(projh, 24,
                                                                WVqh, Vbuf);

  trittention_flash_kernel<<<dim3(NPC, 4, NH), 256, 0, stream>>>(
      projh, Abuf, Vbuf, pm, pl, pacc);

  merge_kernel<<<dim3(NH * 4), 64, 0, stream>>>(pm, pl, pacc, zh);

  gemm_out_kernel<<<dim3(DMODEL / 64, T_SEQ / 128, 1), 128, 0, stream>>>(
      zh, Wouth, bout, out);
}
